// DifferentiableMPO_65128884076657
// MI455X (gfx1250) — matrix operand order repaired and checked by reading, not compiled, not run on hardware
//
#include <hip/hip_runtime.h>
#include <hip/hip_bf16.h>

// ---------------------------------------------------------------------------
// DifferentiableMPO on MI455X (gfx1250, wave32)
//   1) convert L (fp32, 100MB) -> Lbf (bf16, 50MB)           [HBM-bound]
//   2) Sigma_bh = Lbf_bh @ Lbf_bh^T via v_wmma_f32_16x16x32_bf16.
//      B fragments are read as rows of L from global -> 2x b128 loads each
//      (instead of 16x ds_load_u16 gathers). Sigma stored bf16 (50MB,
//      resident in the 192MB L2).
//   3) persistent solver: 1 block / batch, 200 projected-subgradient
//      iterations, w kept in LDS, Sigma streamed from L2 each iteration.
// ---------------------------------------------------------------------------

typedef __bf16 bf16_t;
typedef bf16_t v16bf __attribute__((ext_vector_type(16)));
typedef float  v8f   __attribute__((ext_vector_type(8)));

#define N_    128
#define H_    12
#define GAMMA 5.0f
#define COSTC 1e-3f
#define NITER 200
#define ETA0  0.02f
#define TSTR  136   // padded u16 row stride of transposed LDS tile
                    // (136*2 = 272 = 17*16 bytes: 16B-aligned rows, no
                    //  systematic bank conflicts on the transpose stores)

// bf16 <-> f32 as pure bit ops (no __bf16 scalar arithmetic needed)
__device__ __forceinline__ unsigned short f2bf(float f) {
  union { float f; unsigned u; } c; c.f = f;
  unsigned u = c.u;
  u += 0x7fffu + ((u >> 16) & 1u);           // round-to-nearest-even
  return (unsigned short)(u >> 16);
}
__device__ __forceinline__ float bflo(unsigned u) {
  union { unsigned u; float f; } c; c.u = u << 16; return c.f;
}
__device__ __forceinline__ float bfhi(unsigned u) {
  union { unsigned u; float f; } c; c.u = u & 0xffff0000u; return c.f;
}

// ---------------------------------------------------------------- kernel 1 --
__global__ void __launch_bounds__(256)
k_convert(const float* __restrict__ src, unsigned short* __restrict__ dst, long n) {
  long i = ((long)blockIdx.x * 256 + threadIdx.x) * 8;
  if (i + 8 <= n) {
    const float4* s4 = (const float4*)(src + i);
    float4 a = s4[0], b = s4[1];
    union { unsigned short h[8]; uint4 q; } o;
    o.h[0] = f2bf(a.x); o.h[1] = f2bf(a.y); o.h[2] = f2bf(a.z); o.h[3] = f2bf(a.w);
    o.h[4] = f2bf(b.x); o.h[5] = f2bf(b.y); o.h[6] = f2bf(b.z); o.h[7] = f2bf(b.w);
    *(uint4*)(dst + i) = o.q;
  }
}

// ---------------------------------------------------------------- kernel 2 --
// One block per (b,h): 128x128 Sigma tile. 8 waves, wave mt owns output tile
// rows [16mt,16mt+16); loops nt over 8 column tiles; K-loop = 4 WMMAs of K=32.
__global__ void __launch_bounds__(256)
k_sigma(const unsigned short* __restrict__ Lb, unsigned short* __restrict__ Sg) {
  __shared__ unsigned short ls_t[N_ * TSTR];   // transposed L tile, ~34 KB
  const int bh = blockIdx.x;
  const unsigned short* Lt = Lb + (size_t)bh * N_ * N_;

  // ---- stage L^T into LDS: 2x2 element blocks so stores are b32 ----------
  {
    const unsigned* LtU = (const unsigned*)Lt;       // row = 64 uints
    const int cp  = threadIdx.x & 63;                // column pair 0..63
    const int rp0 = threadIdx.x >> 6;                // row-pair base 0..3
    #pragma unroll
    for (int rb = 0; rb < 16; ++rb) {
      const int rp = rp0 + 4 * rb;                   // row pair 0..63
      const unsigned u0 = LtU[(2 * rp)     * 64 + cp];   // L[2rp  ][2cp,2cp+1]
      const unsigned u1 = LtU[(2 * rp + 1) * 64 + cp];   // L[2rp+1][2cp,2cp+1]
      const unsigned t0 = (u0 & 0xffffu) | (u1 << 16);          // col 2cp
      const unsigned t1 = (u0 >> 16)     | (u1 & 0xffff0000u);  // col 2cp+1
      *(unsigned*)(ls_t + (2 * cp)     * TSTR + 2 * rp) = t0;
      *(unsigned*)(ls_t + (2 * cp + 1) * TSTR + 2 * rp) = t1;
    }
  }

  const int wave = threadIdx.x >> 5;
  const int lane = threadIdx.x & 31;
  const int mt = wave;

  // ---- A fragments for all 4 k-tiles, straight from global ---------------
  // (ISA 7.12.2: lane = row M; lanes<16 hold K 0-7/16-23 of the 32-slice,
  //  lanes>=16 hold K 8-15/24-31)
  union FA { uint4 q[2]; v16bf v; } afr[4];
  {
    const int m    = 16 * mt + (lane & 15);
    const int ksel = (lane >> 4) << 3;               // 0 or 8
    const uint4* lrow = (const uint4*)(Lt + m * N_);
    #pragma unroll
    for (int kt = 0; kt < 4; ++kt) {
      const int ks = 32 * kt + ksel;
      afr[kt].q[0] = lrow[ks >> 3];
      afr[kt].q[1] = lrow[(ks + 16) >> 3];
    }
  }
  __syncthreads();

  unsigned short* out = Sg + (size_t)bh * N_ * N_;
  #pragma unroll 1
  for (int nt = 0; nt < 8; ++nt) {
    v8f c = {0.f, 0.f, 0.f, 0.f, 0.f, 0.f, 0.f, 0.f};
    #pragma unroll
    for (int kt = 0; kt < 4; ++kt) {
      // B fragment: lane holds column n = 16nt + (lane&15) of B = L^T, i.e.
      // L[n][k] for K 0-7/16-23 (lanes<16) or 8-15/24-31 (lanes>=16).
      const int n = 16 * nt + (lane & 15);
      union FB { uint4 q[2]; v16bf v; } bfr;
      const uint4* bp = (const uint4*)(Lt + n * N_ + 32 * kt + ((lane >> 4) << 3));
      bfr.q[0] = bp[0];
      bfr.q[1] = bp[2];
      c = __builtin_amdgcn_wmma_f32_16x16x32_bf16(
              false, afr[kt].v, false, bfr.v, (short)0, c, false, false);
    }
    // C/D layout: VGPR r, lanes<16 -> M=r, lanes>=16 -> M=r+8; N = lane&15
    const int row0 = 16 * mt + ((lane >> 4) << 3);
    const int col  = 16 * nt + (lane & 15);
    #pragma unroll
    for (int r = 0; r < 8; ++r)
      out[(size_t)(row0 + r) * N_ + col] = f2bf(c[r]);
  }
}

// ---------------------------------------------------------------- kernel 3 --
// Persistent solver: block = batch b, 256 threads (8 waves). w lives in LDS
// for all 200 iterations; Sigma (bf16, L2-resident) is the only global read.
__global__ void __launch_bounds__(256)
k_solve(const float* __restrict__ mu, const float* __restrict__ wprev,
        const unsigned short* __restrict__ Sg, float* __restrict__ out) {
  __shared__ float w_s[H_][N_];
  __shared__ float v_s[H_][N_];     // holds y, then v = w - eta*grad
  __shared__ float mu_s[H_][N_];
  __shared__ float wp_s[N_];

  const int b   = blockIdx.x;
  const int tid = threadIdx.x;

  for (int e = tid; e < H_ * N_; e += 256) {
    ((float*)w_s)[e]  = 1.0f / N_;
    ((float*)mu_s)[e] = mu[(size_t)b * H_ * N_ + e];
  }
  if (tid < N_) wp_s[tid] = wprev[(size_t)b * N_ + tid];
  __syncthreads();

  const int g    = tid >> 6;        // 4 groups x 64 threads; group g owns h=3g..3g+2
  const int t64  = tid & 63;        // 2 Sigma columns per thread (coalesced uint loads)
  const int wave = tid >> 5;
  const int lane = tid & 31;
  const unsigned* SgU = (const unsigned*)(Sg + (size_t)b * H_ * N_ * N_);

  for (int k = 0; k < NITER; ++k) {
    const float eta = ETA0 / sqrtf((float)(k + 1));

    // ---- Phase 1: y_h = Sigma_h * w_h  (Jacobi: uses old w) --------------
    #pragma unroll 1
    for (int hh = 0; hh < 3; ++hh) {
      const int h = 3 * g + hh;
      const unsigned* S = SgU + (size_t)h * (N_ * N_ / 2);
      float y0 = 0.f, y1 = 0.f;
      #pragma unroll 4
      for (int j = 0; j < N_; ++j) {
        const unsigned u = S[j * (N_ / 2) + t64];   // 2 bf16, coalesced
        const float wj = w_s[h][j];                 // LDS broadcast
        y0 = fmaf(bflo(u), wj, y0);
        y1 = fmaf(bfhi(u), wj, y1);
      }
      v_s[h][2 * t64]     = y0;
      v_s[h][2 * t64 + 1] = y1;
    }
    __syncthreads();

    // ---- Phase 2: v = w - eta * (-mu + 2*gamma*y + cost*(s - s_next)) ----
    #pragma unroll
    for (int r = 0; r < 6; ++r) {
      const int e = tid + 256 * r;
      const int h = e >> 7, i = e & 127;
      const float wc = w_s[h][i];
      const float wm = (h == 0) ? wp_s[i] : w_s[h - 1][i];
      const float s  = (wc > wm) ? 1.f : ((wc < wm) ? -1.f : 0.f);
      float sn = 0.f;
      if (h < H_ - 1) {
        const float wn = w_s[h + 1][i];
        sn = (wn > wc) ? 1.f : ((wn < wc) ? -1.f : 0.f);
      }
      const float grad = -mu_s[h][i] + 2.f * GAMMA * v_s[h][i] + COSTC * (s - sn);
      v_s[h][i] = wc - eta * grad;
    }
    __syncthreads();

    // ---- Phase 3: exact simplex projection (Michelot), one wave per h ----
    #pragma unroll 1
    for (int hh = 0; hh < 2; ++hh) {
      const int h = wave + 8 * hh;
      if (h < H_) {
        float v0 = v_s[h][lane],      v1 = v_s[h][lane + 32];
        float v2 = v_s[h][lane + 64], v3 = v_s[h][lane + 96];
        float a0 = 1.f, a1 = 1.f, a2 = 1.f, a3 = 1.f;
        float theta = 0.f;
        for (int it = 0; it < 24; ++it) {
          float S = a0 * v0 + a1 * v1 + a2 * v2 + a3 * v3;
          float C = a0 + a1 + a2 + a3;
          #pragma unroll
          for (int off = 16; off; off >>= 1) {
            S += __shfl_xor(S, off, 32);
            C += __shfl_xor(C, off, 32);
          }
          theta = (S - 1.0f) / C;
          const float n0 = v0 > theta ? 1.f : 0.f, n1 = v1 > theta ? 1.f : 0.f;
          const float n2 = v2 > theta ? 1.f : 0.f, n3 = v3 > theta ? 1.f : 0.f;
          const bool ch = (n0 != a0) || (n1 != a1) || (n2 != a2) || (n3 != a3);
          a0 = n0; a1 = n1; a2 = n2; a3 = n3;
          if (!__any(ch)) break;                     // converged (exact proj)
        }
        w_s[h][lane]      = fmaxf(v0 - theta, 0.f);
        w_s[h][lane + 32] = fmaxf(v1 - theta, 0.f);
        w_s[h][lane + 64] = fmaxf(v2 - theta, 0.f);
        w_s[h][lane + 96] = fmaxf(v3 - theta, 0.f);
      }
    }
    __syncthreads();
  }

  for (int e = tid; e < H_ * N_; e += 256)
    out[(size_t)b * H_ * N_ + e] = ((float*)w_s)[e];
}

// ---------------------------------------------------------------------------
extern "C" void kernel_launch(void* const* d_in, const int* in_sizes, int n_in,
                              void* d_out, int out_size, void* d_ws, size_t ws_size,
                              hipStream_t stream) {
  const float* mu = (const float*)d_in[0];   // (B,H,N) fp32
  const float* L  = (const float*)d_in[1];   // (B,H,N,N) fp32
  const float* wp = (const float*)d_in[2];   // (B,N) fp32
  float* out = (float*)d_out;                // (B,H,N) fp32

  const long nL = (long)in_sizes[1];                 // B*H*N*N
  const int  B  = in_sizes[2] / N_;                  // 128
  const int  BH = (int)(nL / ((long)N_ * N_));       // B*H = 1536

  unsigned short* Lb = (unsigned short*)d_ws;        // 50 MB bf16 L
  unsigned short* Sg = Lb + nL;                      // 50 MB bf16 Sigma

  const long cblocks = (nL / 8 + 255) / 256;
  k_convert<<<dim3((unsigned)cblocks), 256, 0, stream>>>(L, Lb, nL);
  k_sigma  <<<dim3((unsigned)BH),      256, 0, stream>>>(Lb, Sg);
  k_solve  <<<dim3((unsigned)B),       256, 0, stream>>>(mu, wp, Sg, out);
}
